// RobertaCrfForNer_33930241638501
// MI455X (gfx1250) — matrix operand order repaired and checked by reading, not compiled, not run on hardware
//
#include <hip/hip_runtime.h>
#include <math.h>
#include <stdint.h>

// ---------------- WMMA types ----------------
typedef __attribute__((ext_vector_type(16))) __bf16 bf16x16;
typedef __attribute__((ext_vector_type(8)))  float  f32x8;

#define WMMA_BF16(a, b, c) \
    __builtin_amdgcn_wmma_f32_16x16x32_bf16(false, (a), false, (b), (short)0, (c), false, false)

static __device__ __forceinline__ f32x8 zero8() {
    f32x8 z = {0.f,0.f,0.f,0.f,0.f,0.f,0.f,0.f};
    return z;
}

// Model dims
#define BB  16
#define TT  512
#define HH  768
#define NHD 12
#define DHD 64
#define FFD 3072
#define LL  9
#define BT  (BB*TT)

// ---------------- CDNA5 async global->LDS copy ----------------
#if defined(__has_builtin)
#  if __has_builtin(__builtin_amdgcn_global_load_async_to_lds_b128)
#    define HAS_ASYNC_LDS 1
#  endif
#endif
#ifndef HAS_ASYNC_LDS
#  define HAS_ASYNC_LDS 0
#endif

typedef int v4i __attribute__((vector_size(16)));

static __device__ __forceinline__ void copy16_g2l(const __bf16* g, __bf16* l)
{
#if HAS_ASYNC_LDS
    __builtin_amdgcn_global_load_async_to_lds_b128((v4i*)g, (v4i*)l, 0, 0);
#else
    *(uint4*)l = *(const uint4*)g;   // fallback: synchronous 16-byte copy
#endif
}

static __device__ __forceinline__ void wait_async_le6()
{
#if HAS_ASYNC_LDS
    asm volatile("s_wait_asynccnt 0x6" ::: "memory");
#endif
}
static __device__ __forceinline__ void wait_async_0()
{
#if HAS_ASYNC_LDS
    asm volatile("s_wait_asynccnt 0x0" ::: "memory");
#endif
}

// =====================================================================
// GEMM: C[M,N] = act(A[M,K] * W[K,N] + bias[N])
// A: bf16 [M,K] row-major.  Wt: bf16 [N,K] (pre-transposed weight).
// Async global->LDS staging, double-buffered. Block 256 thr (8 waves),
// tile 128(M) x 64(N), BK = 64; each wave does 2x2 WMMA 16x16 tiles.
// obf: store bf16 (else fp32).  gelu: exact GELU epilogue.
// =====================================================================
#define GBM 128
#define GBN 64
#define GBK 64
#define LDA 72   // padded bf16 stride (rows are 8 x 16B chunks + pad)
#define LDB 72

__global__ __launch_bounds__(256) void gemm_bf16_kernel(
    const __bf16* __restrict__ A, const __bf16* __restrict__ Wt,
    const float* __restrict__ bias, void* __restrict__ Cout,
    int M, int N, int Kd, int obf, int gelu)
{
    __shared__ __align__(16) __bf16 As[2][GBM * LDA];
    __shared__ __align__(16) __bf16 Bs[2][GBN * LDB];

    const int tid  = threadIdx.x;
    const int lane = tid & 31;
    const int wid  = tid >> 5;
    const int wm   = wid & 3;   // 4 waves along M
    const int wn   = wid >> 2;  // 2 waves along N
    const int lm   = lane & 15;
    const int kh   = lane >> 4;

    const int m0 = blockIdx.x * GBM;
    const int n0 = blockIdx.y * GBN;
    const int S  = Kd / GBK;

    f32x8 acc[2][2];
    acc[0][0] = zero8(); acc[0][1] = zero8();
    acc[1][0] = zero8(); acc[1][1] = zero8();

    // ---- stage issue: 6 x 16B async copies per thread ----
    auto issue_stage = [&](int s) {
        const int buf = s & 1;
        const int k0  = s * GBK;
        #pragma unroll
        for (int e = 0; e < 4; ++e) {            // A tile: 128 rows x 8 chunks
            const int idx = tid + e * 256;
            const int row = idx >> 3;
            const int ch  = idx & 7;
            copy16_g2l(A + (size_t)(m0 + row) * Kd + k0 + ch * 8,
                       &As[buf][row * LDA + ch * 8]);
        }
        #pragma unroll
        for (int e = 0; e < 2; ++e) {            // B tile: 64 rows x 8 chunks
            const int idx = tid + e * 256;
            const int n   = idx >> 3;
            const int ch  = idx & 7;
            copy16_g2l(Wt + (size_t)(n0 + n) * Kd + k0 + ch * 8,
                       &Bs[buf][n * LDB + ch * 8]);
        }
    };

    issue_stage(0);
    for (int s = 0; s < S; ++s) {
        if (s + 1 < S) { issue_stage(s + 1); wait_async_le6(); }
        else           { wait_async_0(); }
        __syncthreads();

        const int buf = s & 1;
        #pragma unroll
        for (int ks = 0; ks < 2; ++ks) {
            bf16x16 af[2], bf[2];
            #pragma unroll
            for (int mt = 0; mt < 2; ++mt) {
                const __bf16* p = &As[buf][(wm * 32 + mt * 16 + lm) * LDA + ks * 32 + kh * 16];
                #pragma unroll
                for (int j = 0; j < 8; ++j) { af[mt][j] = p[j]; af[mt][8 + j] = p[8 + j]; }
            }
            #pragma unroll
            for (int nt = 0; nt < 2; ++nt) {
                const __bf16* p = &Bs[buf][(wn * 32 + nt * 16 + lm) * LDB + ks * 32 + kh * 16];
                #pragma unroll
                for (int i = 0; i < 16; ++i) bf[nt][i] = p[i];
            }
            #pragma unroll
            for (int mt = 0; mt < 2; ++mt)
                #pragma unroll
                for (int nt = 0; nt < 2; ++nt)
                    acc[mt][nt] = WMMA_BF16(af[mt], bf[nt], acc[mt][nt]);
        }
        __syncthreads();
    }

    // ---- epilogue: C layout M = r + 8*kh, N = lm ----
    #pragma unroll
    for (int mt = 0; mt < 2; ++mt) {
        #pragma unroll
        for (int nt = 0; nt < 2; ++nt) {
            #pragma unroll
            for (int r = 0; r < 8; ++r) {
                const int row = m0 + wm * 32 + mt * 16 + kh * 8 + r;
                const int col = n0 + wn * 32 + nt * 16 + lm;
                float v = acc[mt][nt][r] + bias[col];
                if (gelu) v = 0.5f * v * (1.0f + erff(v * 0.70710678118f));
                if (obf) ((__bf16*)Cout)[(size_t)row * N + col] = (__bf16)v;
                else     ((float*)Cout)[(size_t)row * N + col]  = v;
            }
        }
    }
}

// =====================================================================
// Weight transpose + convert: W[K,N] fp32 -> Wt[N,K] bf16. 32x32 tiles.
// =====================================================================
__global__ __launch_bounds__(256) void transpose_cvt_kernel(
    const float* __restrict__ W, __bf16* __restrict__ Wt, int K, int N)
{
    __shared__ float tile[32][33];
    const int n0 = blockIdx.x * 32;
    const int k0 = blockIdx.y * 32;
    const int tx = threadIdx.x & 31;
    const int ty = threadIdx.x >> 5;
    #pragma unroll
    for (int e = 0; e < 4; ++e)
        tile[ty + e * 8][tx] = W[(size_t)(k0 + ty + e * 8) * N + n0 + tx];
    __syncthreads();
    #pragma unroll
    for (int e = 0; e < 4; ++e)
        Wt[(size_t)(n0 + ty + e * 8) * K + k0 + tx] = (__bf16)tile[tx][ty + e * 8];
}

// =====================================================================
// Flash-style attention on bf16 Q/K/V. grid = (T/64, NH, B), block 128.
// Scores scaled by 1/8 post-WMMA; bf16 ctx output.
// =====================================================================
__global__ __launch_bounds__(128) void attn_kernel(
    const __bf16* __restrict__ Q, const __bf16* __restrict__ K,
    const __bf16* __restrict__ V, const int* __restrict__ msk,
    __bf16* __restrict__ ctx)
{
    const int w    = threadIdx.x >> 5;
    const int lane = threadIdx.x & 31;
    const int lm   = lane & 15;
    const int kh   = lane >> 4;
    const int hh   = blockIdx.y;
    const int b    = blockIdx.z;
    const int qbase = blockIdx.x * 64 + w * 16;

    __shared__ __bf16 Ps[4][16 * 64];

    bf16x16 qf[2];
    {
        const __bf16* qrow = Q + ((size_t)(b * TT + qbase + lm)) * HH + hh * DHD;
        #pragma unroll
        for (int ks = 0; ks < 2; ++ks)
            #pragma unroll
            for (int g = 0; g < 2; ++g) {
                const __bf16* p = qrow + ks * 32 + kh * 16 + g * 8;
                #pragma unroll
                for (int j = 0; j < 8; ++j) qf[ks][g * 8 + j] = p[j];
            }
    }

    float mrow[8], lrow[8];
    f32x8 oacc[4];
    #pragma unroll
    for (int r = 0; r < 8; ++r) { mrow[r] = -3.0e38f; lrow[r] = 0.f; }
    #pragma unroll
    for (int nt = 0; nt < 4; ++nt) oacc[nt] = zero8();

    for (int kc = 0; kc < 8; ++kc) {
        const int kbase = kc * 64;

        f32x8 sacc[4];
        #pragma unroll
        for (int nt = 0; nt < 4; ++nt) sacc[nt] = zero8();
        #pragma unroll
        for (int nt = 0; nt < 4; ++nt) {
            const __bf16* kp = K + (size_t)(b * TT + kbase + nt * 16 + lm) * HH + hh * DHD;
            #pragma unroll
            for (int ks = 0; ks < 2; ++ks) {
                bf16x16 bf;
                const __bf16* p = kp + ks * 32 + kh * 16;
                #pragma unroll
                for (int i = 0; i < 16; ++i) bf[i] = p[i];
                sacc[nt] = WMMA_BF16(qf[ks], bf, sacc[nt]);
            }
        }

        float biasv[4];
        #pragma unroll
        for (int nt = 0; nt < 4; ++nt)
            biasv[nt] = (msk[b * TT + kbase + nt * 16 + lm] != 0) ? 0.f : -1.0e9f;

        #pragma unroll
        for (int r = 0; r < 8; ++r) {
            float rmax = -3.0e38f;
            #pragma unroll
            for (int nt = 0; nt < 4; ++nt) {
                float s = sacc[nt][r] * 0.125f + biasv[nt];
                sacc[nt][r] = s;
                rmax = fmaxf(rmax, s);
            }
            #pragma unroll
            for (int off = 1; off < 16; off <<= 1)
                rmax = fmaxf(rmax, __shfl_xor(rmax, off, 32));
            const float mnew = fmaxf(mrow[r], rmax);
            const float corr = expf(mrow[r] - mnew);
            float rsum = 0.f;
            #pragma unroll
            for (int nt = 0; nt < 4; ++nt) {
                float p = expf(sacc[nt][r] - mnew);
                sacc[nt][r] = p;
                rsum += p;
            }
            #pragma unroll
            for (int off = 1; off < 16; off <<= 1)
                rsum += __shfl_xor(rsum, off, 32);
            lrow[r] = lrow[r] * corr + rsum;
            mrow[r] = mnew;
            #pragma unroll
            for (int nt = 0; nt < 4; ++nt) oacc[nt][r] *= corr;
            #pragma unroll
            for (int nt = 0; nt < 4; ++nt)
                Ps[w][(r + 8 * kh) * 64 + nt * 16 + lm] = (__bf16)sacc[nt][r];
        }
        __syncthreads();

        #pragma unroll
        for (int ks = 0; ks < 2; ++ks) {
            bf16x16 pf;
            #pragma unroll
            for (int g = 0; g < 2; ++g)
                #pragma unroll
                for (int j = 0; j < 8; ++j)
                    pf[g * 8 + j] = Ps[w][lm * 64 + ks * 32 + kh * 16 + g * 8 + j];
            #pragma unroll
            for (int nt = 0; nt < 4; ++nt) {
                bf16x16 vf;
                const int d = nt * 16 + lm;
                #pragma unroll
                for (int i = 0; i < 16; ++i) {
                    const int key = kbase + ks * 32 + kh * 16 + i;
                    vf[i] = V[(size_t)(b * TT + key) * HH + hh * DHD + d];
                }
                oacc[nt] = WMMA_BF16(pf, vf, oacc[nt]);
            }
        }
        __syncthreads();
    }

    #pragma unroll
    for (int r = 0; r < 8; ++r) {
        const float inv = 1.f / lrow[r];
        const int t = qbase + r + 8 * kh;
        #pragma unroll
        for (int nt = 0; nt < 4; ++nt) {
            const int d = nt * 16 + lm;
            ctx[(size_t)(b * TT + t) * HH + hh * DHD + d] = (__bf16)(oacc[nt][r] * inv);
        }
    }
}

// =====================================================================
// out = LN(x + res); also emits bf16 copy for downstream WMMA GEMMs.
// =====================================================================
__global__ __launch_bounds__(256) void add_ln_kernel(
    const float* __restrict__ x, const float* __restrict__ res,
    const float* __restrict__ g, const float* __restrict__ bta,
    float* __restrict__ out, __bf16* __restrict__ outb)
{
    const int row = blockIdx.x;
    const int tid = threadIdx.x;
    float vals[3];
    float s = 0.f, s2 = 0.f;
    #pragma unroll
    for (int e = 0; e < 3; ++e) {
        const int c = tid + e * 256;
        float v = x[(size_t)row * HH + c];
        if (res) v += res[(size_t)row * HH + c];
        vals[e] = v; s += v; s2 += v * v;
    }
    #pragma unroll
    for (int off = 16; off > 0; off >>= 1) {
        s  += __shfl_down(s, off, 32);
        s2 += __shfl_down(s2, off, 32);
    }
    __shared__ float rs[8], rs2[8];
    __shared__ float mu_s, rstd_s;
    if ((tid & 31) == 0) { rs[tid >> 5] = s; rs2[tid >> 5] = s2; }
    __syncthreads();
    if (tid == 0) {
        float ts = 0.f, ts2 = 0.f;
        #pragma unroll
        for (int i = 0; i < 8; ++i) { ts += rs[i]; ts2 += rs2[i]; }
        const float mu = ts / (float)HH;
        const float var = ts2 / (float)HH - mu * mu;
        mu_s = mu; rstd_s = rsqrtf(var + 1e-5f);
    }
    __syncthreads();
    const float mu = mu_s, rstd = rstd_s;
    #pragma unroll
    for (int e = 0; e < 3; ++e) {
        const int c = tid + e * 256;
        const float r = (vals[e] - mu) * rstd * g[c] + bta[c];
        out[(size_t)row * HH + c] = r;
        if (outb) outb[(size_t)row * HH + c] = (__bf16)r;
    }
}

// =====================================================================
// pos[b,t] = cumsum(mask)[t] * mask[t] + 1  (PAD = 1)
// =====================================================================
__global__ void pos_kernel(const int* __restrict__ msk, int* __restrict__ pos)
{
    const int idx = blockIdx.x * blockDim.x + threadIdx.x;
    if (idx >= BT) return;
    const int b = idx / TT, t = idx % TT;
    int c = 0;
    for (int j = 0; j <= t; ++j) c += msk[b * TT + j];
    pos[idx] = c * msk[idx] + 1;
}

__global__ __launch_bounds__(256) void embed_kernel(
    const int* __restrict__ ids, const int* __restrict__ pos,
    const float* __restrict__ we, const float* __restrict__ pe,
    const float* __restrict__ te, float* __restrict__ h)
{
    const int tok = blockIdx.x;
    const int id = ids[tok];
    const int p  = pos[tok];
    #pragma unroll
    for (int e = 0; e < 3; ++e) {
        const int c = threadIdx.x + e * 256;
        h[(size_t)tok * HH + c] =
            we[(size_t)id * HH + c] + pe[(size_t)p * HH + c] + te[c];
    }
}

// =====================================================================
// emissions[tok, l] = dot(h[tok], cls_w[:, l]) + cls_b[l]
// =====================================================================
__global__ __launch_bounds__(288) void cls_kernel(
    const float* __restrict__ h, const float* __restrict__ W,
    const float* __restrict__ bias, float* __restrict__ emis)
{
    const int tok = blockIdx.x;
    const int l    = threadIdx.x >> 5;
    const int lane = threadIdx.x & 31;
    float s = 0.f;
    for (int c = lane; c < HH; c += 32)
        s += h[(size_t)tok * HH + c] * W[c * LL + l];
    #pragma unroll
    for (int off = 16; off > 0; off >>= 1) s += __shfl_down(s, off, 32);
    if (lane == 0) emis[(size_t)tok * LL + l] = s + bias[l];
}

// =====================================================================
// CRF log-likelihood, one block (1 wave) per batch element.
// =====================================================================
__global__ __launch_bounds__(32) void crf_kernel(
    const float* __restrict__ emis, const int* __restrict__ labels,
    const int* __restrict__ msk, const float* __restrict__ start,
    const float* __restrict__ endv, const float* __restrict__ trans,
    float* __restrict__ llh)
{
    const int b = blockIdx.x;
    const int j = threadIdx.x;
    const float* e = emis + (size_t)b * TT * LL;
    __shared__ float sc[LL], nx[LL];
    __shared__ float num_s;

    if (j == 0) {
        int t0 = labels[b * TT]; if (t0 == -100) t0 = 0;
        float num = start[t0] + e[t0];
        int prev = t0;
        for (int t = 1; t < TT; ++t) {
            int tg = labels[b * TT + t]; if (tg == -100) tg = 0;
            if (msk[b * TT + t] != 0)
                num += trans[prev * LL + tg] + e[(size_t)t * LL + tg];
            prev = tg;
        }
        int cnt = 0;
        for (int t = 0; t < TT; ++t) cnt += (msk[b * TT + t] != 0);
        int li = cnt - 1; if (li < 0) li = 0;
        int lt = labels[b * TT + li]; if (lt == -100) lt = 0;
        num += endv[lt];
        num_s = num;
    }
    if (j < LL) sc[j] = start[j] + e[j];
    __syncthreads();

    for (int t = 1; t < TT; ++t) {
        if (j < LL) {
            float tmp[LL];
            float m = -3.0e38f;
            #pragma unroll
            for (int i = 0; i < LL; ++i) {
                float v = sc[i] + trans[i * LL + j];
                tmp[i] = v; m = fmaxf(m, v);
            }
            float s = 0.f;
            #pragma unroll
            for (int i = 0; i < LL; ++i) s += expf(tmp[i] - m);
            float n = m + logf(s) + e[(size_t)t * LL + j];
            nx[j] = (msk[b * TT + t] != 0) ? n : sc[j];
        }
        __syncthreads();
        if (j < LL) sc[j] = nx[j];
        __syncthreads();
    }

    if (j == 0) {
        float m = -3.0e38f;
        #pragma unroll
        for (int i = 0; i < LL; ++i) m = fmaxf(m, sc[i] + endv[i]);
        float s = 0.f;
        #pragma unroll
        for (int i = 0; i < LL; ++i) s += expf(sc[i] + endv[i] - m);
        llh[b] = num_s - (m + logf(s));
    }
}

__global__ void finalize_kernel(const float* __restrict__ llh, float* __restrict__ out)
{
    if (threadIdx.x == 0 && blockIdx.x == 0) {
        float s = 0.f;
        for (int i = 0; i < BB; ++i) s += llh[i];
        out[0] = -s / (float)BB;
    }
}

// =====================================================================
// Host-side launcher
// =====================================================================
static inline void launch_gemm(const __bf16* A, const __bf16* Wt, const float* bias,
                               void* C, int M, int N, int Kd, int obf, int gelu,
                               hipStream_t stream)
{
    dim3 grid(M / GBM, N / GBN);
    gemm_bf16_kernel<<<grid, 256, 0, stream>>>(A, Wt, bias, C, M, N, Kd, obf, gelu);
}

static inline void launch_tr(const float* W, __bf16* Wt, int K, int N, hipStream_t stream)
{
    dim3 grid(N / 32, K / 32);
    transpose_cvt_kernel<<<grid, 256, 0, stream>>>(W, Wt, K, N);
}

extern "C" void kernel_launch(void* const* d_in, const int* in_sizes, int n_in,
                              void* d_out, int out_size, void* d_ws, size_t ws_size,
                              hipStream_t stream)
{
    (void)in_sizes; (void)n_in; (void)out_size; (void)ws_size;

    // ---- inputs (setup_inputs() insertion order, params recursively) ----
    const int*   input_ids = (const int*)d_in[0];
    const int*   attn_mask = (const int*)d_in[1];
    const int*   labels    = (const int*)d_in[2];
    const float* word_emb  = (const float*)d_in[3];
    const float* pos_emb   = (const float*)d_in[4];
    const float* type_emb  = (const float*)d_in[5];
    const float* emb_ln_g  = (const float*)d_in[6];
    const float* emb_ln_b  = (const float*)d_in[7];
    const int LBASE = 8;   // 12 layers x 16 tensors
    const float* cls_w     = (const float*)d_in[LBASE + 12 * 16 + 0];
    const float* cls_b     = (const float*)d_in[LBASE + 12 * 16 + 1];
    const float* crf_start = (const float*)d_in[LBASE + 12 * 16 + 2];
    const float* crf_end   = (const float*)d_in[LBASE + 12 * 16 + 3];
    const float* crf_trans = (const float*)d_in[LBASE + 12 * 16 + 4];

    // ---- workspace carve (256B aligned chunks) ----
    char* wp = (char*)d_ws;
    auto alloc = [&](size_t bytes) -> void* {
        void* r = (void*)wp;
        wp += (bytes + 255) & ~(size_t)255;
        return r;
    };
    float*  h    = (float*) alloc((size_t)BT * HH * 4);
    float*  t1   = (float*) alloc((size_t)BT * HH * 4);
    __bf16* hb   = (__bf16*)alloc((size_t)BT * HH * 2);
    __bf16* qb   = (__bf16*)alloc((size_t)BT * HH * 2);
    __bf16* kb   = (__bf16*)alloc((size_t)BT * HH * 2);
    __bf16* vb   = (__bf16*)alloc((size_t)BT * HH * 2);
    __bf16* cxb  = (__bf16*)alloc((size_t)BT * HH * 2);
    __bf16* fb   = (__bf16*)alloc((size_t)BT * FFD * 2);
    __bf16* wt_q = (__bf16*)alloc((size_t)HH * HH * 2);
    __bf16* wt_k = (__bf16*)alloc((size_t)HH * HH * 2);
    __bf16* wt_v = (__bf16*)alloc((size_t)HH * HH * 2);
    __bf16* wt_o = (__bf16*)alloc((size_t)HH * HH * 2);
    __bf16* wt_f1= (__bf16*)alloc((size_t)HH * FFD * 2);
    __bf16* wt_f2= (__bf16*)alloc((size_t)FFD * HH * 2);
    float*  emis = (float*) alloc((size_t)BT * LL * 4);
    float*  llh  = (float*) alloc(BB * 4);
    int*    pos  = (int*)   alloc((size_t)BT * 4);

    // ---- embeddings ----
    pos_kernel<<<(BT + 255) / 256, 256, 0, stream>>>(attn_mask, pos);
    embed_kernel<<<BT, 256, 0, stream>>>(input_ids, pos, word_emb, pos_emb, type_emb, h);
    add_ln_kernel<<<BT, 256, 0, stream>>>(h, nullptr, emb_ln_g, emb_ln_b, h, hb);

    // ---- encoder layers ----
    for (int L = 0; L < 12; ++L) {
        const float* const* P = (const float* const*)(d_in + LBASE + L * 16);
        const float* q_w = P[0],  *q_b = P[1];
        const float* k_w = P[2],  *k_b = P[3];
        const float* v_w = P[4],  *v_b = P[5];
        const float* o_w = P[6],  *o_b = P[7];
        const float* ln1_g = P[8], *ln1_b = P[9];
        const float* fc1_w = P[10], *fc1_b = P[11];
        const float* fc2_w = P[12], *fc2_b = P[13];
        const float* ln2_g = P[14], *ln2_b = P[15];

        launch_tr(q_w,  wt_q,  HH, HH,  stream);
        launch_tr(k_w,  wt_k,  HH, HH,  stream);
        launch_tr(v_w,  wt_v,  HH, HH,  stream);
        launch_tr(o_w,  wt_o,  HH, HH,  stream);
        launch_tr(fc1_w, wt_f1, HH, FFD, stream);
        launch_tr(fc2_w, wt_f2, FFD, HH, stream);

        launch_gemm(hb, wt_q, q_b, qb, BT, HH, HH, 1, 0, stream);
        launch_gemm(hb, wt_k, k_b, kb, BT, HH, HH, 1, 0, stream);
        launch_gemm(hb, wt_v, v_b, vb, BT, HH, HH, 1, 0, stream);

        attn_kernel<<<dim3(TT / 64, NHD, BB), 128, 0, stream>>>(qb, kb, vb, attn_mask, cxb);

        launch_gemm(cxb, wt_o, o_b, t1, BT, HH, HH, 0, 0, stream);
        add_ln_kernel<<<BT, 256, 0, stream>>>(h, t1, ln1_g, ln1_b, h, hb);

        launch_gemm(hb, wt_f1, fc1_b, fb, BT, FFD, HH, 1, 1, stream);   // bf16 + GELU
        launch_gemm(fb, wt_f2, fc2_b, t1, BT, HH, FFD, 0, 0, stream);
        add_ln_kernel<<<BT, 256, 0, stream>>>(h, t1, ln2_g, ln2_b, h, hb);
    }

    // ---- classifier + CRF ----
    cls_kernel<<<BT, 288, 0, stream>>>(h, cls_w, cls_b, emis);
    crf_kernel<<<BB, 32, 0, stream>>>(emis, labels, attn_mask,
                                      crf_start, crf_end, crf_trans, llh);
    finalize_kernel<<<1, 32, 0, stream>>>(llh, (float*)d_out);
}
